// GCNDirectional_9594956939369
// MI455X (gfx1250) — compile-verified, matrix operand order checked by reading, not run on hardware
//
#include <hip/hip_runtime.h>
#include <cstddef>

#define DD 64
#define LN_EPS 1e-5f

typedef __bf16 bf16_t;
typedef __attribute__((ext_vector_type(16))) __bf16 v16bf;
typedef __attribute__((ext_vector_type(8)))  float  v8f;

// Guaranteed hardware fp32 atomic add (no CAS fallback), device scope so it is
// coherent across WGPs in L2.
__device__ __forceinline__ void atomic_add_f32_dev(float* addr, float val) {
    asm volatile("global_atomic_add_f32 %0, %1, off scope:SCOPE_DEV"
                 :: "v"(addr), "v"(val)
                 : "memory");
}

// ---------------- degree / norm ----------------
__global__ void k_deg_init(float* __restrict__ deg, int N) {
    int n = blockIdx.x * blockDim.x + threadIdx.x;
    if (n < N) deg[n] = 1.0f;            // self-loop contributes 1
}

__global__ void k_deg_count(const int* __restrict__ row, float* __restrict__ deg, int E) {
    int e = blockIdx.x * blockDim.x + threadIdx.x;
    if (e < E) atomic_add_f32_dev(&deg[row[e]], 1.0f);
}

__global__ void k_dinv(float* __restrict__ d, int N) {
    int n = blockIdx.x * blockDim.x + threadIdx.x;
    if (n < N) d[n] = rsqrtf(d[n]);      // deg >= 1 always
}

// ---------------- GEMM: H = X (N x 64) @ W (64 x 64), bf16 WMMA ----------------
// One wave per 16-row strip. W staged to LDS pre-swizzled into WMMA B-fragment
// order: Wl[frag][lane][e], frag = kkIdx*4 + colTile, so each lane's fragment
// is 32 contiguous bytes -> 2x ds_load_b128 instead of 16x ds_load_u16.
__global__ void __launch_bounds__(256)
k_gemm(const float* __restrict__ X, const float* __restrict__ W,
       float* __restrict__ H, int nstrips, int N) {
    __shared__ __align__(32) bf16_t Wl[8 * 32 * 16];   // 8 frags x 32 lanes x 16 elems
    int tid = threadIdx.x;
    // stage + swizzle: element i -> (frag, lane, e) -> W[k*64 + n]
    for (int i = tid; i < 8 * 32 * 16; i += 256) {
        int e     = i & 15;
        int ln    = (i >> 4) & 31;
        int frag  = i >> 9;
        int kkIdx = frag >> 2;          // 0..1  (K chunk of 32)
        int ct    = frag & 3;           // 0..3  (16-col tile)
        int half  = ln >> 4;
        int l15   = ln & 15;
        int k     = kkIdx * 32 + half * 8 + e + (e < 8 ? 0 : 8);
        int ncol  = ct * 16 + l15;
        Wl[i] = (bf16_t)W[k * DD + ncol];
    }
    __syncthreads();

    int wave  = tid >> 5;
    int lane  = tid & 31;
    int strip = blockIdx.x * 8 + wave;
    if (strip >= nstrips) return;

    int half = lane >> 4;       // 0: lanes 0-15, 1: lanes 16-31
    int l15  = lane & 15;
    int rowBase = strip * 16;
    int rload   = rowBase + l15;
    if (rload > N - 1) rload = N - 1;    // clamp for a possible ragged tail
    const float4* xrow = (const float4*)(X + (size_t)rload * DD);  // 16B-aligned rows

    v8f acc[4] = {};

#pragma unroll
    for (int kkIdx = 0; kkIdx < 2; ++kkIdx) {
        // A fragment: elements 0..7 = K[kk+half*8 .. +7], 8..15 = K[+16 .. +23]
        int kbase = kkIdx * 32 + half * 8;        // multiple of 4 -> float4 index
        float4 f0 = xrow[(kbase)      >> 2];
        float4 f1 = xrow[(kbase + 4)  >> 2];
        float4 f2 = xrow[(kbase + 16) >> 2];
        float4 f3 = xrow[(kbase + 20) >> 2];
        v16bf a;
        a[0]  = (bf16_t)f0.x; a[1]  = (bf16_t)f0.y; a[2]  = (bf16_t)f0.z; a[3]  = (bf16_t)f0.w;
        a[4]  = (bf16_t)f1.x; a[5]  = (bf16_t)f1.y; a[6]  = (bf16_t)f1.z; a[7]  = (bf16_t)f1.w;
        a[8]  = (bf16_t)f2.x; a[9]  = (bf16_t)f2.y; a[10] = (bf16_t)f2.z; a[11] = (bf16_t)f2.w;
        a[12] = (bf16_t)f3.x; a[13] = (bf16_t)f3.y; a[14] = (bf16_t)f3.z; a[15] = (bf16_t)f3.w;
#pragma unroll
        for (int ct = 0; ct < 4; ++ct) {
            const v16bf* bp = (const v16bf*)&Wl[((kkIdx * 4 + ct) * 32 + lane) * 16];
            v16bf b = *bp;                         // 2x ds_load_b128
            acc[ct] = __builtin_amdgcn_wmma_f32_16x16x32_bf16(
                false, a, false, b, (short)0, acc[ct], false, false);
        }
    }

    // C/D layout: element r -> row (rowBase + half*8 + r), col (ct*16 + l15)
#pragma unroll
    for (int ct = 0; ct < 4; ++ct) {
#pragma unroll
        for (int r = 0; r < 8; ++r) {
            int m = rowBase + half * 8 + r;
            if (m < N) H[(size_t)m * DD + ct * 16 + l15] = acc[ct][r];
        }
    }
}

// ---------------- acc = x + bias + h * dinv^2  (skip + bias + self-loop) ----------------
__global__ void k_init_acc(const float* __restrict__ xin, const float* __restrict__ h,
                           const float* __restrict__ dinv, const float* __restrict__ bias,
                           float* __restrict__ acc, int N) {
    int t = blockIdx.x * blockDim.x + threadIdx.x;
    if (t >= N * DD) return;
    int n = t >> 6, d = t & 63;
    float di = dinv[n];
    acc[t] = xin[t] + bias[d] + h[t] * di * di;
}

// ---------------- edge scatter: acc[row] += norm * h[col] ----------------
// One wave per edge; lane handles dims lane and lane+32.
__global__ void k_scatter(const int* __restrict__ ei, const float* __restrict__ h,
                          const float* __restrict__ dinv, float* __restrict__ acc, int E) {
    int t = blockIdx.x * blockDim.x + threadIdx.x;
    int e = t >> 5;
    if (e >= E) return;
    int lane = t & 31;
    int r = ei[e];
    int c = ei[E + e];
    float w = dinv[r] * dinv[c];
    const float* hc = h + (size_t)c * DD;
    float m0 = hc[lane]      * w;
    float m1 = hc[32 + lane] * w;
    float* ar = acc + (size_t)r * DD;
    atomic_add_f32_dev(ar + lane,      m0);
    atomic_add_f32_dev(ar + 32 + lane, m1);
}

// ---------------- relu + LayerNorm (one wave32 per node) ----------------
__global__ void __launch_bounds__(256)
k_post_ln(const float* __restrict__ acc, const float* __restrict__ g,
          const float* __restrict__ b, float* __restrict__ xout, int N) {
    int wave = threadIdx.x >> 5;
    int lane = threadIdx.x & 31;
    int n = blockIdx.x * 8 + wave;
    if (n >= N) return;
    const float* an = acc + (size_t)n * DD;
    float v0 = fmaxf(an[lane], 0.0f);
    float v1 = fmaxf(an[32 + lane], 0.0f);
    float s = v0 + v1;
    float q = v0 * v0 + v1 * v1;
#pragma unroll
    for (int off = 16; off >= 1; off >>= 1) {
        s += __shfl_xor(s, off, 32);
        q += __shfl_xor(q, off, 32);
    }
    float mu   = s * (1.0f / 64.0f);
    float var  = q * (1.0f / 64.0f) - mu * mu;    // biased var, like jnp.var
    float rstd = rsqrtf(var + LN_EPS);
    float* xo = xout + (size_t)n * DD;
    xo[lane]      = (v0 - mu) * rstd * g[lane]      + b[lane];
    xo[32 + lane] = (v1 - mu) * rstd * g[32 + lane] + b[32 + lane];
}

// ---------------- final layer: emb = acc ; x = relu(acc) ----------------
__global__ void k_post_final(const float* __restrict__ acc, float* __restrict__ out, int N) {
    int t = blockIdx.x * blockDim.x + threadIdx.x;
    if (t >= N * DD) return;
    float v = acc[t];
    out[t] = v;                                   // emb
    out[(size_t)N * DD + t] = fmaxf(v, 0.0f);     // x (relu, no LN after last layer)
}

extern "C" void kernel_launch(void* const* d_in, const int* in_sizes, int n_in,
                              void* d_out, int out_size, void* d_ws, size_t ws_size,
                              hipStream_t stream) {
    const float* x  = (const float*)d_in[0];
    const int*   ei = (const int*)  d_in[1];
    const float* Ws = (const float*)d_in[2];
    const float* bs = (const float*)d_in[3];
    const float* lg = (const float*)d_in[4];
    const float* lb = (const float*)d_in[5];
    (void)n_in; (void)out_size; (void)ws_size;

    const int N = in_sizes[0] / DD;
    const int E = in_sizes[1] / 2;
    const int L = in_sizes[3] / DD;

    // workspace partition (256B aligned)
    char* ws = (char*)d_ws;
    auto take = [&](size_t bytes) {
        char* p = ws;
        ws += (bytes + 255) & ~(size_t)255;
        return p;
    };
    float* dinv = (float*)take((size_t)N * sizeof(float));
    float* h    = (float*)take((size_t)N * DD * sizeof(float));
    float* acc  = (float*)take((size_t)N * DD * sizeof(float));
    float* xbuf = (float*)take((size_t)N * DD * sizeof(float));

    // degrees -> dinv (recomputed every call; no persistent state)
    k_deg_init <<<(N + 255) / 256, 256, 0, stream>>>(dinv, N);
    k_deg_count<<<(E + 255) / 256, 256, 0, stream>>>(ei, dinv, E);
    k_dinv     <<<(N + 255) / 256, 256, 0, stream>>>(dinv, N);

    const int nstrips = (N + 15) / 16;
    const int gemmBlk = (nstrips + 7) / 8;
    const int ndBlk   = (N * DD + 255) / 256;
    const int scatBlk = (int)(((size_t)E * 32 + 255) / 256);
    const int postBlk = (N + 7) / 8;

    const float* xin = x;
    for (int i = 0; i < L; ++i) {
        k_gemm    <<<gemmBlk, 256, 0, stream>>>(xin, Ws + (size_t)i * DD * DD, h, nstrips, N);
        k_init_acc<<<ndBlk,   256, 0, stream>>>(xin, h, dinv, bs + (size_t)i * DD, acc, N);
        k_scatter <<<scatBlk, 256, 0, stream>>>(ei, h, dinv, acc, E);
        if (i < L - 1) {
            k_post_ln<<<postBlk, 256, 0, stream>>>(acc, lg + (size_t)i * DD,
                                                   lb + (size_t)i * DD, xbuf, N);
            xin = xbuf;
        } else {
            k_post_final<<<ndBlk, 256, 0, stream>>>(acc, (float*)d_out, N);
        }
    }
}
